// MoEMLP_53111565582398
// MI455X (gfx1250) — matrix operand order repaired and checked by reading, not compiled, not run on hardware
//
#include <hip/hip_runtime.h>
#include <hip/hip_bf16.h>
#include <math.h>

#define S_TOK 8192
#define C_DIM 1024
#define H_DIM 4096
#define E_NUM 8
#define CAP   2560   // ceil(1.25 * 8192 * 2 / 8)

typedef __attribute__((ext_vector_type(16))) __bf16 v16bf;
typedef __attribute__((ext_vector_type(8)))  float  v8f;
typedef unsigned int v4u __attribute__((ext_vector_type(4)));
typedef int v8i __attribute__((ext_vector_type(8)));
typedef int v4i __attribute__((ext_vector_type(4)));

union Frag16 {
    v16bf v;
    uint4 q[2];
    unsigned u[8];
};

__device__ __forceinline__ unsigned pack_bf16(float a, float b) {
    // RNE float->bf16, a -> [15:0], b -> [31:16]
    unsigned ua = __float_as_uint(a);
    unsigned ub = __float_as_uint(b);
    ua += 0x7FFFu + ((ua >> 16) & 1u);
    ub += 0x7FFFu + ((ub >> 16) & 1u);
    return (ua >> 16) | (ub & 0xFFFF0000u);
}

__device__ __forceinline__ unsigned short bf16_bits(float a) {
    unsigned ua = __float_as_uint(a);
    ua += 0x7FFFu + ((ua >> 16) & 1u);
    return (unsigned short)(ua >> 16);
}

// ---- Tensor Data Mover helpers (2-D tile, D# per CDNA5 ISA) --
// group0: [1:0]=count, [63:32]=lds_addr, [120:64]=global_addr, [127:126]=type(2)
// group1: [17:16]=data_size, [79:48]=tensor_dim0, [111:80]=tensor_dim1,
//         [127:112]=tile_dim0, [143:128]=tile_dim1, [207:160]=tensor_dim0_stride
__device__ __forceinline__ void tdm_desc(v4u& g0, v8i& g1, unsigned lds_off,
                                         unsigned long long gaddr, unsigned dsz_code,
                                         unsigned tile_d0, unsigned tile_d1,
                                         unsigned long long d0_stride_elems) {
    g0[0] = 1u;                                        // count=1 (valid, user mode)
    g0[1] = lds_off;                                   // LDS byte address
    g0[2] = (unsigned)gaddr;                           // global addr [31:0]
    g0[3] = ((unsigned)(gaddr >> 32) & 0x1FFFFFFu) | 0x80000000u;  // [56:32] | type=2
    g1[0] = (int)(dsz_code << 16);                     // workgroup_mask=0, data_size
    g1[1] = (int)(tile_d0 << 16);                      // tensor_dim0 = tile_d0 (lo16)
    g1[2] = (int)(tile_d1 << 16);                      // dim0 hi=0 | tensor_dim1 lo16
    g1[3] = (int)(tile_d0 << 16);                      // dim1 hi=0 | tile_dim0
    g1[4] = (int)tile_d1;                              // tile_dim1 | tile_dim2=0
    g1[5] = (int)(unsigned)d0_stride_elems;            // tensor_dim0_stride [31:0]
    g1[6] = (int)(unsigned)(d0_stride_elems >> 32);    // stride hi | dim1_stride lo=0
    g1[7] = 0;
}

__device__ __forceinline__ void tdm_load_2d(unsigned lds_off, unsigned long long gaddr,
                                            unsigned dsz_code, unsigned tile_d0,
                                            unsigned tile_d1, unsigned long long stride) {
    v4u g0; v8i g1; v4i gz = {}; v8i gz8 = {};
    tdm_desc(g0, g1, lds_off, gaddr, dsz_code, tile_d0, tile_d1, stride);
    __builtin_amdgcn_tensor_load_to_lds(g0, g1, gz, gz, gz8, 0);
}

__device__ __forceinline__ void tdm_store_2d(unsigned lds_off, unsigned long long gaddr,
                                             unsigned dsz_code, unsigned tile_d0,
                                             unsigned tile_d1, unsigned long long stride) {
    v4u g0; v8i g1; v4i gz = {}; v8i gz8 = {};
    tdm_desc(g0, g1, lds_off, gaddr, dsz_code, tile_d0, tile_d1, stride);
    __builtin_amdgcn_tensor_store_from_lds(g0, g1, gz, gz, gz8, 0);
}

// ---------------------------------------------------------------- init
__global__ void moe_init(float* __restrict__ y, int* __restrict__ token_buf,
                         float* __restrict__ gate_buf, float* __restrict__ probsum,
                         int* __restrict__ cnt) {
    int idx = blockIdx.x * blockDim.x + threadIdx.x;
    if (idx < S_TOK * C_DIM) y[idx] = 0.0f;
    if (idx < E_NUM * CAP) { token_buf[idx] = 0; gate_buf[idx] = 0.0f; }
    if (idx < E_NUM) { probsum[idx] = 0.0f; cnt[idx] = 0; }
}

// ---------------------------------------------------------------- router
// One wave32 per token; 8 waves (8 tokens) per 256-thread block.
__global__ __launch_bounds__(256) void moe_router(
        const float* __restrict__ x, const float* __restrict__ rw,
        const float* __restrict__ rb, float* __restrict__ gate_se,
        float* __restrict__ probsum) {
    __shared__ float wlds[C_DIM * E_NUM];   // 32 KB router weights
    __shared__ float spsum[E_NUM];
    int t = threadIdx.x;
    if (t < E_NUM) spsum[t] = 0.0f;
    for (int i = t; i < C_DIM * E_NUM; i += 256) wlds[i] = rw[i];
    __syncthreads();

    int wave = t >> 5, lane = t & 31;
    int s = blockIdx.x * 8 + wave;
    const float* xr = x + (size_t)s * C_DIM;

    float l[E_NUM] = {};
    for (int c = lane; c < C_DIM; c += 32) {
        float xv = xr[c];
        const float* wr = &wlds[c * E_NUM];
#pragma unroll
        for (int e = 0; e < E_NUM; ++e) l[e] += xv * wr[e];
    }
#pragma unroll
    for (int e = 0; e < E_NUM; ++e)
        for (int off = 16; off >= 1; off >>= 1)
            l[e] += __shfl_xor(l[e], off, 32);

    if (lane == 0) {
        float mx = l[0] + rb[0];
#pragma unroll
        for (int e = 0; e < E_NUM; ++e) { l[e] += rb[e]; mx = fmaxf(mx, l[e]); }
        float p[E_NUM], sum = 0.0f;
#pragma unroll
        for (int e = 0; e < E_NUM; ++e) { p[e] = __expf(l[e] - mx); sum += p[e]; }
        float inv = 1.0f / sum;
#pragma unroll
        for (int e = 0; e < E_NUM; ++e) p[e] *= inv;
        // top-2, ties -> lowest index (matches lax.top_k)
        int e0 = 0;
        for (int e = 1; e < E_NUM; ++e) if (p[e] > p[e0]) e0 = e;
        int e1 = (e0 == 0) ? 1 : 0;
        for (int e = 0; e < E_NUM; ++e) if (e != e0 && p[e] > p[e1]) e1 = e;
#pragma unroll
        for (int e = 0; e < E_NUM; ++e) {
            float g = (e == e0) ? p[e0] : ((e == e1) ? p[e1] : -1.0f);
            gate_se[(size_t)e * S_TOK + s] = g;
            atomicAdd(&spsum[e], p[e]);
        }
    }
    __syncthreads();
    if (t < E_NUM) atomicAdd(&probsum[t], spsum[t]);
}

// ---------------------------------------------------------------- rank/assign
// rank[s] = #{s': g'>g} + #{s'<s: g'==g}  == argsort(argsort(-gate)) (stable)
__global__ __launch_bounds__(256) void moe_assign(
        const float* __restrict__ gate_se, int* __restrict__ token_buf,
        float* __restrict__ gate_buf, int* __restrict__ cnt) {
    __shared__ float g[S_TOK];              // 32 KB
    int e = blockIdx.y;
    int t = threadIdx.x;
    const float* ge = gate_se + (size_t)e * S_TOK;
    for (int i = t; i < S_TOK; i += 256) g[i] = ge[i];
    __syncthreads();

    int s = blockIdx.x * 256 + t;
    float gs = g[s];
    if (gs >= 0.0f) {
        int rank = 0;
        for (int sp = 0; sp < S_TOK; ++sp) {
            float gp = g[sp];
            rank += (gp > gs) || ((gp == gs) && (sp < s));
        }
        if (rank < CAP) {
            token_buf[e * CAP + rank] = s;
            gate_buf[e * CAP + rank] = gs;
            atomicAdd(&cnt[e], 1);
        }
    }
}

// ---------------------------------------------------------------- GEMM1
// h[e, m, n] = gelu( sum_k x[tok[m], k] * w1[e, k, n] + b1[e, n] ), bf16 out
// Output tile staged in LDS, written to hbuf by one TENSOR_STORE_FROM_LDS.
__global__ __launch_bounds__(256) void moe_gemm1(
        const float* __restrict__ x, const float* __restrict__ w1,
        const float* __restrict__ b1, const int* __restrict__ token_buf,
        unsigned short* __restrict__ hbuf) {
    __shared__ __align__(16) char arena[32 * 1024];
    unsigned* shA = (unsigned*)arena;                   // 8 KB: 128 rows x 32 k bf16
    unsigned* shB = (unsigned*)(arena + 8192);          // 8 KB: 128 n x 32 k, transposed
    int* tokL = (int*)(arena + 16384);                  // 512 B (dead before shOut use)
    unsigned short* shOut = (unsigned short*)arena;     // 32 KB: 128 x 128 h tile

    int e = blockIdx.z;
    int m0 = blockIdx.y * 128;
    int n0 = blockIdx.x * 128;
    int t = threadIdx.x;
    if (t < 128) tokL[t] = token_buf[e * CAP + m0 + t];
    __syncthreads();

    int lane = t & 31, wave = t >> 5;
    int wm = wave >> 1, wn = wave & 1;
    int mbase = wm * 32, nbase = wn * 64;
    int half = lane >> 4, ln = lane & 15;

    int arow = t >> 1, ad = (t & 1) * 8;    // A fill: row, dword offset
    int kp = t >> 4, nc = (t & 15) * 8;     // B fill: k-pair, n chunk

    const float* w1e = w1 + (size_t)e * C_DIM * H_DIM;
    v8f acc[2][4] = {};

    for (int k0 = 0; k0 < C_DIM; k0 += 32) {
        // stage A (gathered token rows), fp32 -> bf16
        {
            const float* src = x + (size_t)tokL[arow] * C_DIM + k0 + ad * 2;
            float4 f0 = ((const float4*)src)[0];
            float4 f1 = ((const float4*)src)[1];
            float4 f2 = ((const float4*)src)[2];
            float4 f3 = ((const float4*)src)[3];
            unsigned* dst = &shA[arow * 16 + ad];
            dst[0] = pack_bf16(f0.x, f0.y); dst[1] = pack_bf16(f0.z, f0.w);
            dst[2] = pack_bf16(f1.x, f1.y); dst[3] = pack_bf16(f1.z, f1.w);
            dst[4] = pack_bf16(f2.x, f2.y); dst[5] = pack_bf16(f2.z, f2.w);
            dst[6] = pack_bf16(f3.x, f3.y); dst[7] = pack_bf16(f3.z, f3.w);
        }
        // stage B transposed: shB[n][kpair] = (w1[k][n], w1[k+1][n])
        {
            const float* s0 = w1e + (size_t)(k0 + 2 * kp) * H_DIM + n0 + nc;
            const float* s1 = s0 + H_DIM;
            float4 a0 = ((const float4*)s0)[0];
            float4 a1 = ((const float4*)s0)[1];
            float4 c0 = ((const float4*)s1)[0];
            float4 c1 = ((const float4*)s1)[1];
            shB[(nc + 0) * 16 + kp] = pack_bf16(a0.x, c0.x);
            shB[(nc + 1) * 16 + kp] = pack_bf16(a0.y, c0.y);
            shB[(nc + 2) * 16 + kp] = pack_bf16(a0.z, c0.z);
            shB[(nc + 3) * 16 + kp] = pack_bf16(a0.w, c0.w);
            shB[(nc + 4) * 16 + kp] = pack_bf16(a1.x, c1.x);
            shB[(nc + 5) * 16 + kp] = pack_bf16(a1.y, c1.y);
            shB[(nc + 6) * 16 + kp] = pack_bf16(a1.z, c1.z);
            shB[(nc + 7) * 16 + kp] = pack_bf16(a1.w, c1.w);
            if (k0 + 32 < C_DIM) __builtin_prefetch(s0 + 32 * H_DIM, 0, 1);
        }
        __syncthreads();

        Frag16 a[2], b[4];
#pragma unroll
        for (int fm = 0; fm < 2; ++fm) {
            int m = mbase + fm * 16 + ln;
            a[fm].q[0] = *(const uint4*)&shA[m * 16 + half * 4];
            a[fm].q[1] = *(const uint4*)&shA[m * 16 + 8 + half * 4];
        }
#pragma unroll
        for (int fn = 0; fn < 4; ++fn) {
            int n = nbase + fn * 16 + ln;
            b[fn].q[0] = *(const uint4*)&shB[n * 16 + half * 4];
            b[fn].q[1] = *(const uint4*)&shB[n * 16 + 8 + half * 4];
        }
#pragma unroll
        for (int fm = 0; fm < 2; ++fm)
#pragma unroll
            for (int fn = 0; fn < 4; ++fn)
                acc[fm][fn] = __builtin_amdgcn_wmma_f32_16x16x32_bf16(
                    false, a[fm].v, false, b[fn].v, (short)0, acc[fm][fn],
                    false, false);
        __syncthreads();
    }

    // epilogue: bias + exact gelu, stage bf16 tile in LDS
    const float* b1e = b1 + (size_t)e * H_DIM;
#pragma unroll
    for (int fm = 0; fm < 2; ++fm)
#pragma unroll
        for (int fn = 0; fn < 4; ++fn) {
            int nl = nbase + fn * 16 + ln;
            float bias = b1e[n0 + nl];
#pragma unroll
            for (int v = 0; v < 8; ++v) {
                int ml = mbase + fm * 16 + half * 8 + v;
                float val = acc[fm][fn][v] + bias;
                val = 0.5f * val * (1.0f + erff(val * 0.70710678118654752f));
                shOut[ml * 128 + nl] = bf16_bits(val);
            }
        }
    __syncthreads();
    // one TDM store: 128x128 bf16 tile, row stride H
    if (t == 0) {
        unsigned long long ga = (unsigned long long)(size_t)hbuf +
            ((size_t)(e * CAP + m0) * H_DIM + n0) * 2ull;
        tdm_store_2d((unsigned)(size_t)shOut, ga, 1u, 128u, 128u,
                     (unsigned long long)H_DIM);
    }
    __builtin_amdgcn_s_wait_tensorcnt(0);
}

// ---------------------------------------------------------------- GEMM2
// y[tok[m], n] += gate[m] * ( sum_k h[e,m,k] * w2[e,k,n] + b2[e,n] )
// A tile (already bf16, contiguous rows) fetched by TENSOR_LOAD_TO_LDS.
__global__ __launch_bounds__(256) void moe_gemm2(
        const unsigned short* __restrict__ hbuf, const float* __restrict__ w2,
        const float* __restrict__ b2, const int* __restrict__ token_buf,
        const float* __restrict__ gate_buf, float* __restrict__ y) {
    __shared__ __align__(16) unsigned shA[128 * 16];
    __shared__ __align__(16) unsigned shB[128 * 16];
    __shared__ int tokL[128];
    __shared__ float gL[128];

    int e = blockIdx.z;
    int m0 = blockIdx.y * 128;
    int n0 = blockIdx.x * 128;
    int t = threadIdx.x;
    if (t < 128) {
        tokL[t] = token_buf[e * CAP + m0 + t];
        gL[t] = gate_buf[e * CAP + m0 + t];
    }
    __syncthreads();

    int lane = t & 31, wave = t >> 5;
    int wm = wave >> 1, wn = wave & 1;
    int mbase = wm * 32, nbase = wn * 64;
    int half = lane >> 4, ln = lane & 15;

    int kp = t >> 4, nc = (t & 15) * 8;

    const float* w2e = w2 + (size_t)e * H_DIM * C_DIM;
    unsigned shA_off = (unsigned)(size_t)&shA[0];
    unsigned long long habase = (unsigned long long)(size_t)hbuf +
        (size_t)(e * CAP + m0) * H_DIM * 2ull;
    v8f acc[2][4] = {};

    for (int k0 = 0; k0 < H_DIM; k0 += 32) {
        // stage A via TDM: 128 rows x 32 bf16, row stride H -> contiguous LDS
        if (t == 0) {
            tdm_load_2d(shA_off, habase + (unsigned long long)k0 * 2ull,
                        1u, 32u, 128u, (unsigned long long)H_DIM);
        }
        // stage B transposed (fp32 -> bf16)
        {
            const float* s0 = w2e + (size_t)(k0 + 2 * kp) * C_DIM + n0 + nc;
            const float* s1 = s0 + C_DIM;
            float4 a0 = ((const float4*)s0)[0];
            float4 a1 = ((const float4*)s0)[1];
            float4 c0 = ((const float4*)s1)[0];
            float4 c1 = ((const float4*)s1)[1];
            shB[(nc + 0) * 16 + kp] = pack_bf16(a0.x, c0.x);
            shB[(nc + 1) * 16 + kp] = pack_bf16(a0.y, c0.y);
            shB[(nc + 2) * 16 + kp] = pack_bf16(a0.z, c0.z);
            shB[(nc + 3) * 16 + kp] = pack_bf16(a0.w, c0.w);
            shB[(nc + 4) * 16 + kp] = pack_bf16(a1.x, c1.x);
            shB[(nc + 5) * 16 + kp] = pack_bf16(a1.y, c1.y);
            shB[(nc + 6) * 16 + kp] = pack_bf16(a1.z, c1.z);
            shB[(nc + 7) * 16 + kp] = pack_bf16(a1.w, c1.w);
            if (k0 + 32 < H_DIM) __builtin_prefetch(s0 + 32 * C_DIM, 0, 1);
        }
        __builtin_amdgcn_s_wait_tensorcnt(0);   // TENSORcnt==0 for non-issuing waves
        __syncthreads();

        Frag16 a[2], b[4];
#pragma unroll
        for (int fm = 0; fm < 2; ++fm) {
            int m = mbase + fm * 16 + ln;
            a[fm].q[0] = *(const uint4*)&shA[m * 16 + half * 4];
            a[fm].q[1] = *(const uint4*)&shA[m * 16 + 8 + half * 4];
        }
#pragma unroll
        for (int fn = 0; fn < 4; ++fn) {
            int n = nbase + fn * 16 + ln;
            b[fn].q[0] = *(const uint4*)&shB[n * 16 + half * 4];
            b[fn].q[1] = *(const uint4*)&shB[n * 16 + 8 + half * 4];
        }
#pragma unroll
        for (int fm = 0; fm < 2; ++fm)
#pragma unroll
            for (int fn = 0; fn < 4; ++fn)
                acc[fm][fn] = __builtin_amdgcn_wmma_f32_16x16x32_bf16(
                    false, a[fm].v, false, b[fn].v, (short)0, acc[fm][fn],
                    false, false);
        __syncthreads();
    }

    const float* b2e = b2 + (size_t)e * C_DIM;
#pragma unroll
    for (int fm = 0; fm < 2; ++fm)
#pragma unroll
        for (int fn = 0; fn < 4; ++fn) {
            int gn = n0 + nbase + fn * 16 + ln;
            float bias = b2e[gn];
#pragma unroll
            for (int v = 0; v < 8; ++v) {
                int ml = mbase + fm * 16 + half * 8 + v;
                float val = (acc[fm][fn][v] + bias) * gL[ml];
                atomicAdd(&y[(size_t)tokL[ml] * C_DIM + gn], val);
            }
        }
}

// ---------------------------------------------------------------- aux loss
__global__ void moe_aux(const float* __restrict__ probsum,
                        const int* __restrict__ cnt, float* __restrict__ out_aux) {
    if (threadIdx.x == 0 && blockIdx.x == 0) {
        float tsum = 0.0f;
        for (int e = 0; e < E_NUM; ++e) tsum += (float)cnt[e];
        float a = 0.0f;
        for (int e = 0; e < E_NUM; ++e)
            a += (probsum[e] / (float)S_TOK) * ((float)cnt[e] / (tsum + 1e-9f));
        *out_aux = a * (float)E_NUM;
    }
}

// ---------------------------------------------------------------- launch
extern "C" void kernel_launch(void* const* d_in, const int* in_sizes, int n_in,
                              void* d_out, int out_size, void* d_ws, size_t ws_size,
                              hipStream_t stream) {
    const float* x  = (const float*)d_in[0];
    const float* rw = (const float*)d_in[1];
    const float* rb = (const float*)d_in[2];
    const float* w1 = (const float*)d_in[3];
    const float* b1 = (const float*)d_in[4];
    const float* w2 = (const float*)d_in[5];
    const float* b2 = (const float*)d_in[6];

    float* y   = (float*)d_out;                      // [S, C]
    float* aux = y + (size_t)S_TOK * C_DIM;          // scalar

    char* ws = (char*)d_ws;
    size_t off = 0;
    unsigned short* hbuf = (unsigned short*)(ws + off);
    off += (size_t)E_NUM * CAP * H_DIM * sizeof(unsigned short);   // 168 MB
    float* gate_se = (float*)(ws + off); off += (size_t)E_NUM * S_TOK * 4;
    int* token_buf = (int*)(ws + off);   off += (size_t)E_NUM * CAP * 4;
    float* gate_buf = (float*)(ws + off); off += (size_t)E_NUM * CAP * 4;
    float* probsum = (float*)(ws + off); off += 64;
    int* cnt = (int*)(ws + off);

    moe_init<<<(S_TOK * C_DIM + 255) / 256, 256, 0, stream>>>(
        y, token_buf, gate_buf, probsum, cnt);
    moe_router<<<S_TOK / 8, 256, 0, stream>>>(x, rw, rb, gate_se, probsum);
    moe_assign<<<dim3(S_TOK / 256, E_NUM), 256, 0, stream>>>(
        gate_se, token_buf, gate_buf, cnt);
    moe_gemm1<<<dim3(H_DIM / 128, CAP / 128, E_NUM), 256, 0, stream>>>(
        x, w1, b1, token_buf, hbuf);
    moe_gemm2<<<dim3(C_DIM / 128, CAP / 128, E_NUM), 256, 0, stream>>>(
        hbuf, w2, b2, token_buf, gate_buf, y);
    moe_aux<<<1, 1, 0, stream>>>(probsum, cnt, aux);
}
